// DiffusionOrderingNetwork_1322849927183
// MI455X (gfx1250) — matrix operand order repaired and checked by reading, not compiled, not run on hardware
//
#include <hip/hip_runtime.h>
#include <math.h>

#define N_NODES 100000
#define N_EDGES 1600000
#define NREL    5
#define HDIM    64
#define KABS    10000
#define NTYPE   11
#define NBASE   2
#define RM      (NREL + 1)   // 5 relations + root as a 6th "relation"

typedef _Float16 half_t;
typedef __attribute__((ext_vector_type(16))) _Float16 v16h;
typedef __attribute__((ext_vector_type(8)))  _Float16 v8h;
typedef __attribute__((ext_vector_type(8)))  float    v8f;

// ---------------- setup kernels ----------------

__global__ void k_zero(int* cnt, int* absorbed, float* denom) {
  int i = blockIdx.x * blockDim.x + threadIdx.x;
  if (i < N_NODES * NREL) cnt[i] = 0;
  if (i < N_NODES) absorbed[i] = 0;
  if (i == 0) *denom = 0.f;
}

__global__ void k_absorb(const int* __restrict__ node_order, int* __restrict__ absorbed) {
  int k = blockIdx.x * blockDim.x + threadIdx.x;
  if (k < KABS) absorbed[node_order[k]] = 1;
}

__global__ void k_count(const int* __restrict__ dst, const int* __restrict__ et,
                        int* __restrict__ cnt) {
  int e = blockIdx.x * blockDim.x + threadIdx.x;
  if (e < N_EDGES) atomicAdd(&cnt[dst[e] * NREL + et[e]], 1);
}

__global__ void k_embed(const int* __restrict__ x, const float* __restrict__ emb,
                        float* __restrict__ hf) {
  int i = blockIdx.x * blockDim.x + threadIdx.x;
  if (i >= N_NODES * HDIM) return;
  int n = i >> 6, j = i & 63;
  hf[i] = emb[x[n] * HDIM + j];
}

__global__ void k_pe(const int* __restrict__ node_order, float* __restrict__ hf) {
  int i = blockIdx.x * blockDim.x + threadIdx.x;
  if (i >= KABS * HDIM) return;
  int k = i >> 6, j = i & 63;
  atomicAdd(&hf[(size_t)node_order[k] * HDIM + j], sinf((float)(k + 1)));
}

__global__ void k_tohalf(const float* __restrict__ hf, half_t* __restrict__ hh) {
  int i = blockIdx.x * blockDim.x + threadIdx.x;
  if (i < N_NODES * HDIM) hh[i] = (half_t)hf[i];
}

// ---------------- weight build: Wt[r][out][in] (f16, transposed) ----------------

__global__ void k_makeW(const float* __restrict__ bases, const float* __restrict__ comp,
                        const float* __restrict__ root, half_t* __restrict__ Wt) {
  int i = blockIdx.x * blockDim.x + threadIdx.x;      // RM * 64 * 64
  if (i >= RM * HDIM * HDIM) return;
  int r  = i / (HDIM * HDIM);
  int o  = (i / HDIM) % HDIM;   // output column
  int ii = i % HDIM;            // input row (K dim)
  float w;
  if (r < NREL) {
    w = 0.f;
    for (int b = 0; b < NBASE; ++b)
      w += comp[r * NBASE + b] * bases[((size_t)b * HDIM + ii) * HDIM + o];
  } else {
    w = root[ii * HDIM + o];
  }
  Wt[((size_t)r * HDIM + o) * HDIM + ii] = (half_t)w;
}

// ---------------- WMMA GEMM: xwh[n][r][o] = sum_i hh[n][i] * W[r][i][o] ----------------
// One wave computes a 16-node x 64-col tile for one (node_tile, r).
// A (16x32 f16): lane row = lane&15, halves split by hi = lane>>4 per ISA layout.
// B (32x16 f16): K values 8*hi..8*hi+7 and 16+8*hi..16+8*hi+7 of Wt row (r*64 + ct*16 + col).

__global__ void k_gemm(const half_t* __restrict__ hh, const half_t* __restrict__ Wt,
                       half_t* __restrict__ xwh) {
  const int waves = (N_NODES / 16) * RM;   // 6250 * 6 = 37500
  int wid = (int)((blockIdx.x * blockDim.x + threadIdx.x) >> 5);
  if (wid >= waves) return;
  int lane = threadIdx.x & 31;
  int r    = wid % RM;
  int n0   = (wid / RM) * 16;
  int row  = lane & 15;     // A row / B-C column
  int hi   = lane >> 4;

  v8f acc[4] = {v8f{}, v8f{}, v8f{}, v8f{}};

  #pragma unroll
  for (int kc = 0; kc < HDIM; kc += 32) {
    const half_t* ap = hh + (size_t)(n0 + row) * HDIM + kc + 8 * hi;
    v8h alo = *(const v8h*)(ap);
    v8h ahi = *(const v8h*)(ap + 16);
    v16h a;
    #pragma unroll
    for (int t = 0; t < 8; ++t) { a[t] = alo[t]; a[t + 8] = ahi[t]; }
    #pragma unroll
    for (int ct = 0; ct < 4; ++ct) {
      const half_t* bp = Wt + ((size_t)(r * HDIM) + ct * 16 + row) * HDIM + kc + 8 * hi;
      v8h blo = *(const v8h*)(bp);
      v8h bhi = *(const v8h*)(bp + 16);
      v16h b;
      #pragma unroll
      for (int t = 0; t < 8; ++t) { b[t] = blo[t]; b[t + 8] = bhi[t]; }
      acc[ct] = __builtin_amdgcn_wmma_f32_16x16x32_f16(
          false, a, false, b, (short)0, acc[ct], false, false);
    }
  }

  // acc[ct][j] -> out element (M = n0 + j + 8*hi, N = ct*16 + row)
  #pragma unroll
  for (int ct = 0; ct < 4; ++ct)
    #pragma unroll
    for (int j = 0; j < 8; ++j)
      xwh[(size_t)(n0 + j + 8 * hi) * (RM * HDIM) + r * HDIM + ct * 16 + row] =
          (half_t)acc[ct][j];
}

// ---------------- aggregation (accumulate directly into next-layer hh, f16) ----------------

__global__ void k_init(const half_t* __restrict__ xwh, const float* __restrict__ bias,
                       half_t* __restrict__ hh) {
  int i = blockIdx.x * blockDim.x + threadIdx.x;
  if (i >= N_NODES * HDIM) return;
  int n = i >> 6, j = i & 63;
  hh[i] = (half_t)((float)xwh[(size_t)n * (RM * HDIM) + NREL * HDIM + j] + bias[j]);
}

// One thread per (edge, f16-pair): packed f16 atomic add (GLOBAL_ATOMIC_PK_ADD_F16)
// halves scatter bytes and atomic op count vs scalar f32 atomics.
__global__ void k_edge(const half_t* __restrict__ xwh, const int* __restrict__ src,
                       const int* __restrict__ dst, const int* __restrict__ et,
                       const int* __restrict__ cnt, half_t* __restrict__ hh) {
  long long i = (long long)blockIdx.x * blockDim.x + threadIdx.x;
  if (i >= (long long)N_EDGES * (HDIM / 2)) return;
  int e = (int)(i >> 5), p = (int)(i & 31);       // pair index 0..31
  int t = et[e], s = src[e], d = dst[e];
  int c = cnt[d * NREL + t];
  float norm = 1.f / (float)(c > 1 ? c : 1);
  const half_t* mp = xwh + (size_t)s * (RM * HDIM) + t * HDIM + p * 2;
  float f0 = (float)mp[0] * norm;
  float f1 = (float)mp[1] * norm;
  union { half_t h[2]; unsigned int u; } pk;
  pk.h[0] = (half_t)f0;
  pk.h[1] = (half_t)f1;
  half_t* ap = hh + (size_t)d * HDIM + p * 2;
  asm volatile("global_atomic_pk_add_f16 %0, %1, off"
               :: "v"((unsigned long long)(size_t)ap), "v"(pk.u)
               : "memory");
}

// ---------------- layer 2 (dout = 1) ----------------

__global__ void k_makeW2(const float* __restrict__ bases, const float* __restrict__ comp,
                         const float* __restrict__ root, float* __restrict__ w2) {
  int i = blockIdx.x * blockDim.x + threadIdx.x;
  if (i >= RM * HDIM) return;
  int r = i / HDIM, ii = i % HDIM;
  float w;
  if (r < NREL) {
    w = 0.f;
    for (int b = 0; b < NBASE; ++b) w += comp[r * NBASE + b] * bases[b * HDIM + ii];
  } else {
    w = root[ii];
  }
  w2[i] = w;
}

__global__ void k_l2(const half_t* __restrict__ hh, const float* __restrict__ w2,
                     float* __restrict__ xw2) {
  int i = blockIdx.x * blockDim.x + threadIdx.x;
  if (i >= N_NODES * RM) return;
  int n = i / RM, r = i % RM;
  float s = 0.f;
  #pragma unroll 8
  for (int k = 0; k < HDIM; ++k) s += (float)hh[(size_t)n * HDIM + k] * w2[r * HDIM + k];
  xw2[i] = s;
}

__global__ void k_init2(const float* __restrict__ xw2, const float* __restrict__ bias2,
                        float* __restrict__ out2) {
  int n = blockIdx.x * blockDim.x + threadIdx.x;
  if (n < N_NODES) out2[n] = xw2[n * RM + NREL] + bias2[0];
}

__global__ void k_edge2(const float* __restrict__ xw2, const int* __restrict__ src,
                        const int* __restrict__ dst, const int* __restrict__ et,
                        const int* __restrict__ cnt, float* __restrict__ out2) {
  int e = blockIdx.x * blockDim.x + threadIdx.x;
  if (e >= N_EDGES) return;
  int t = et[e], s = src[e], d = dst[e];
  int c = cnt[d * NREL + t];
  float norm = 1.f / (float)(c > 1 ? c : 1);
  atomicAdd(&out2[d], xw2[s * RM + t] * norm);
}

// ---------------- masked softmax ----------------

__global__ void k_soft1(const float* __restrict__ out2, const int* __restrict__ absorbed,
                        float* __restrict__ out, float* __restrict__ denom) {
  __shared__ float red[256];
  int n = blockIdx.x * blockDim.x + threadIdx.x;
  float e = 0.f;
  if (n < N_NODES) {
    e = absorbed[n] ? 0.f : expf(out2[n]);
    out[n] = e;
  }
  red[threadIdx.x] = e;
  __syncthreads();
  for (int s = 128; s > 0; s >>= 1) {
    if (threadIdx.x < s) red[threadIdx.x] += red[threadIdx.x + s];
    __syncthreads();
  }
  if (threadIdx.x == 0) atomicAdd(denom, red[0]);
}

__global__ void k_soft2(float* __restrict__ out, const float* __restrict__ denom) {
  int n = blockIdx.x * blockDim.x + threadIdx.x;
  if (n < N_NODES) out[n] = out[n] / (*denom);
}

// ---------------- host launch ----------------

static inline int gridFor(long long n, int b) { return (int)((n + b - 1) / b); }

extern "C" void kernel_launch(void* const* d_in, const int* in_sizes, int n_in,
                              void* d_out, int out_size, void* d_ws, size_t ws_size,
                              hipStream_t stream) {
  (void)in_sizes; (void)n_in; (void)out_size; (void)ws_size;

  const int*   x      = (const int*)d_in[0];
  const int*   ei     = (const int*)d_in[1];
  const int*   etype  = (const int*)d_in[2];
  const int*   norder = (const int*)d_in[3];
  const float* emb    = (const float*)d_in[4];
  const float* basesL[3] = {(const float*)d_in[5],  (const float*)d_in[9],  (const float*)d_in[13]};
  const float* compL[3]  = {(const float*)d_in[6],  (const float*)d_in[10], (const float*)d_in[14]};
  const float* rootL[3]  = {(const float*)d_in[7],  (const float*)d_in[11], (const float*)d_in[15]};
  const float* biasL[3]  = {(const float*)d_in[8],  (const float*)d_in[12], (const float*)d_in[16]};
  const int* src = ei;
  const int* dst = ei + N_EDGES;

  // workspace bump allocator (256B aligned)
  char* ws = (char*)d_ws;
  size_t off = 0;
  auto alloc = [&](size_t bytes) -> char* {
    char* p = ws + off;
    off = (off + bytes + 255) & ~(size_t)255;
    return p;
  };
  half_t* hh   = (half_t*)alloc((size_t)N_NODES * HDIM * sizeof(half_t));
  half_t* xwh  = (half_t*)alloc((size_t)N_NODES * RM * HDIM * sizeof(half_t));
  float*  hf   = (float*) alloc((size_t)N_NODES * HDIM * sizeof(float));
  int*    cnt  = (int*)   alloc((size_t)N_NODES * NREL * sizeof(int));
  half_t* Wt   = (half_t*)alloc((size_t)RM * HDIM * HDIM * sizeof(half_t));
  float*  w2   = (float*) alloc((size_t)RM * HDIM * sizeof(float));
  float*  xw2  = (float*) alloc((size_t)N_NODES * RM * sizeof(float));
  float*  out2 = (float*) alloc((size_t)N_NODES * sizeof(float));
  int*    absorbed = (int*)alloc((size_t)N_NODES * sizeof(int));
  float*  denom = (float*)alloc(sizeof(float));
  float*  outf = (float*)d_out;

  const int B = 256;

  k_zero  <<<gridFor((long long)N_NODES * NREL, B), B, 0, stream>>>(cnt, absorbed, denom);
  k_absorb<<<gridFor(KABS, B), B, 0, stream>>>(norder, absorbed);
  k_count <<<gridFor(N_EDGES, B), B, 0, stream>>>(dst, etype, cnt);
  k_embed <<<gridFor((long long)N_NODES * HDIM, B), B, 0, stream>>>(x, emb, hf);
  k_pe    <<<gridFor((long long)KABS * HDIM, B), B, 0, stream>>>(norder, hf);
  k_tohalf<<<gridFor((long long)N_NODES * HDIM, B), B, 0, stream>>>(hf, hh);

  const int gemmWaves  = (N_NODES / 16) * RM;          // 37500 wave tiles
  const int gemmBlocks = gridFor((long long)gemmWaves * 32, B);

  for (int l = 0; l < 2; ++l) {
    k_makeW<<<gridFor(RM * HDIM * HDIM, B), B, 0, stream>>>(basesL[l], compL[l], rootL[l], Wt);
    // gemm reads hh -> xwh; then hh is re-initialized (root+bias) and edge
    // messages are accumulated into it in-place with packed f16 atomics.
    k_gemm <<<gemmBlocks, B, 0, stream>>>(hh, Wt, xwh);
    k_init <<<gridFor((long long)N_NODES * HDIM, B), B, 0, stream>>>(xwh, biasL[l], hh);
    k_edge <<<gridFor((long long)N_EDGES * (HDIM / 2), B), B, 0, stream>>>(xwh, src, dst, etype, cnt, hh);
  }

  // layer 2: dout = 1
  k_makeW2<<<gridFor(RM * HDIM, B), B, 0, stream>>>(basesL[2], compL[2], rootL[2], w2);
  k_l2    <<<gridFor((long long)N_NODES * RM, B), B, 0, stream>>>(hh, w2, xw2);
  k_init2 <<<gridFor(N_NODES, B), B, 0, stream>>>(xw2, biasL[2], out2);
  k_edge2 <<<gridFor(N_EDGES, B), B, 0, stream>>>(xw2, src, dst, etype, cnt, out2);

  // masked softmax over unmasked nodes
  k_soft1<<<gridFor(N_NODES, B), B, 0, stream>>>(out2, absorbed, outf, denom);
  k_soft2<<<gridFor(N_NODES, B), B, 0, stream>>>(outf, denom);
}
